// MultiHead_2310692405963
// MI455X (gfx1250) — compile-verified, matrix operand order checked by reading, not run on hardware
//
#include <hip/hip_runtime.h>

// Problem constants (reference: B=4, S=2048, D=1024, H=16, HD=64)
#define B_  4
#define S_  2048
#define D_  1024
#define H_  16
#define HD_ 64

typedef __attribute__((ext_vector_type(16))) __bf16 v16bf;
typedef __attribute__((ext_vector_type(8)))  float  v8f;

union FragA { v16bf v; unsigned int u[8]; };

__device__ __forceinline__ unsigned short f2bf(float f) {
  unsigned int u = __float_as_uint(f);
  unsigned int r = u + 0x7FFFu + ((u >> 16) & 1u);  // round-to-nearest-even
  return (unsigned short)(r >> 16);
}
__device__ __forceinline__ unsigned short toBF(float x)          { return f2bf(x); }
__device__ __forceinline__ unsigned short toBF(unsigned short x) { return x; }

// K-offset of VGPR j in a 16-bit 16x32 A-fragment / 32x16 B-fragment.
// lanes 0-15 (hi=0): V0-3 -> K={0,2,4,6}, V4-7 -> K={16,18,20,22}; lanes 16-31: +8.
__device__ __forceinline__ int kkOf(int j, int hi) {
  return (hi ? 8 : 0) + ((j < 4) ? 2 * j : 16 + 2 * (j - 4));
}

// Build a bf16 WMMA fragment from LDS. `rc` = row (A) or column (B, stored
// transposed [n][k]); `stride` in elements; `off` = extra K offset.
__device__ __forceinline__ v16bf load_frag_lds(const unsigned short* base,
                                               int stride, int rc, int hi, int off) {
  FragA f;
#pragma unroll
  for (int j = 0; j < 8; ++j)
    f.u[j] = *(const unsigned int*)(base + rc * stride + off + kkOf(j, hi));
  return f.v;
}

// ---------------------------------------------------------------------------
// Tensor Data Mover: 1-D bf16 tile (32 keys x 64 d, contiguous in global) into
// LDS with hardware padding of 1 DWORD per 32 DWORDs -> LDS row stride 66
// elements (bank-conflict free for the B-fragment reads).
// This toolchain has the 6-arg builtin:
//   (u32x4 g0, i32x8 g1, i32x4 g2, i32x4 g3, i32x8 g4, i32 cpol)
// ---------------------------------------------------------------------------
#if defined(__has_builtin)
#if __has_builtin(__builtin_amdgcn_tensor_load_to_lds) && \
    __has_builtin(__builtin_amdgcn_s_wait_tensorcnt)
#define USE_TDM 1
#endif
#endif
#ifndef USE_TDM
#define USE_TDM 0
#endif

#if USE_TDM
typedef __attribute__((ext_vector_type(4))) unsigned int u32x4;
typedef __attribute__((ext_vector_type(8))) int          i32x8;
typedef __attribute__((ext_vector_type(4))) int          i32x4;

__device__ __forceinline__ void tdm_load_k_tile(unsigned lds_byte_addr,
                                                const unsigned short* gsrc) {
  unsigned long long ga = (unsigned long long)(const void*)gsrc;
  const unsigned n = 32 * HD_;  // 2048 bf16 elements (4 KB)
  u32x4 g0;
  g0.x = 1u;                                                  // count=1, user D#
  g0.y = lds_byte_addr;                                       // lds_addr
  g0.z = (unsigned)ga;                                        // global_addr[31:0]
  g0.w = ((unsigned)(ga >> 32) & 0x01FFFFFFu) | (2u << 30);   // addr[56:32]|type=2
  i32x8 g1;
  g1[0] = (int)((1u << 16) |        // data_size = 2 bytes
                (1u << 20) |        // pad_enable
                (4u << 22));        // pad_interval: every 32 DWORDs (+1 DWORD pad)
  g1[1] = (int)((n & 0xFFFFu) << 16);                 // tensor_dim0[15:0]
  g1[2] = (int)(((n >> 16) & 0xFFFFu) | (1u << 16));  // tensor_dim0[31:16], dim1=1
  g1[3] = (int)((n & 0xFFFFu) << 16);                 // tile_dim0 = n
  g1[4] = 0;                                          // tile_dim1/2 unused
  g1[5] = (int)n;                                     // tensor_dim0_stride
  g1[6] = 0;
  g1[7] = 0;
  const i32x4 z4 = {0, 0, 0, 0};
  const i32x8 z8 = {0, 0, 0, 0, 0, 0, 0, 0};
  __builtin_amdgcn_tensor_load_to_lds(g0, g1, z4, z4, z8, 0);
}
#endif

// ---------------------------------------------------------------------------
// Tiled GEMM: C[M=8192, N=1024] = A[M,1024] * W[1024,1024] + bias
// ---------------------------------------------------------------------------
template <typename TA, bool HEADOUT>
__global__ __launch_bounds__(256) void gemm_kernel(const TA* __restrict__ A,
                                                   const float* __restrict__ W,
                                                   const float* __restrict__ bias,
                                                   void* __restrict__ outp) {
  const int K = D_, N = D_;
  __shared__ unsigned short As[128 * 34];  // [row][k], stride 34 (pad)
  __shared__ unsigned short Bs[64 * 34];   // [n][k] transposed, stride 34

  const int tid  = threadIdx.x;
  const int wid  = tid >> 5;
  const int lane = tid & 31;
  const int hi   = lane >> 4;
  const int ln   = lane & 15;
  const int wm   = wid >> 1;  // 0..3 : wave M index
  const int wn   = wid & 1;   // 0..1 : wave N index
  const int blockN = blockIdx.x * 64;
  const int blockM = blockIdx.y * 128;

  const v8f vz = {0.f, 0.f, 0.f, 0.f, 0.f, 0.f, 0.f, 0.f};
  v8f acc[2][2] = {{vz, vz}, {vz, vz}};

  for (int kt = 0; kt < K; kt += 32) {
    // Prefetch next iteration's tiles into cache (global_prefetch_b8).
    if (kt + 32 < K) {
      __builtin_prefetch(A + (size_t)(blockM + (tid & 127)) * K + kt + 32, 0, 1);
      __builtin_prefetch(W + (size_t)(kt + 32 + (tid & 31)) * N + blockN, 0, 1);
    }
    // A tile: 128x32 (16 elems/thread, coalesced)
#pragma unroll
    for (int i = 0; i < 16; ++i) {
      int idx = tid + i * 256;
      int r = idx >> 5, c = idx & 31;
      As[r * 34 + c] = toBF(A[(size_t)(blockM + r) * K + kt + c]);
    }
    // W tile: 32x64 -> stored transposed (8 elems/thread, coalesced)
#pragma unroll
    for (int i = 0; i < 8; ++i) {
      int idx = tid + i * 256;
      int k = idx >> 6, n = idx & 63;
      Bs[n * 34 + k] = f2bf(W[(size_t)(kt + k) * N + blockN + n]);
    }
    __syncthreads();

    v16bf afr[2], bfr[2];
#pragma unroll
    for (int t = 0; t < 2; ++t)
      afr[t] = load_frag_lds(As, 34, wm * 32 + t * 16 + ln, hi, 0);
#pragma unroll
    for (int t = 0; t < 2; ++t)
      bfr[t] = load_frag_lds(Bs, 34, wn * 32 + t * 16 + ln, hi, 0);

#pragma unroll
    for (int i = 0; i < 2; ++i)
#pragma unroll
      for (int j = 0; j < 2; ++j)
        acc[i][j] = __builtin_amdgcn_wmma_f32_16x16x32_bf16(
            false, afr[i], false, bfr[j], (short)0, acc[i][j], false, false);
    __syncthreads();
  }

  // Epilogue. C layout: VGPR r -> row r (lanes 0-15) / r+8 (lanes 16-31); N = ln.
#pragma unroll
  for (int i = 0; i < 2; ++i)
#pragma unroll
    for (int j = 0; j < 2; ++j)
#pragma unroll
      for (int r = 0; r < 8; ++r) {
        int m = blockM + wm * 32 + i * 16 + r + hi * 8;
        int n = blockN + wn * 32 + j * 16 + ln;
        float val = acc[i][j][r] + bias[n];
        if constexpr (HEADOUT) {
          unsigned short* o = (unsigned short*)outp;
          int b = m >> 11, s = m & (S_ - 1);
          int hh = n >> 6, hd = n & 63;
          o[((size_t)(b * H_ + hh) * S_ + s) * HD_ + hd] = f2bf(val);
        } else {
          float* o = (float*)outp;
          o[(size_t)m * N + n] = val;
        }
      }
}

// ---------------------------------------------------------------------------
// Flash attention: grid = (S/128, B*H), 256 threads (8 waves).
// K tile arrives via TDM (tensor_load_to_lds, hardware-padded stride 66);
// V tile loaded transposed by the threads. Online softmax in fp32.
// ---------------------------------------------------------------------------
__global__ __launch_bounds__(256) void flash_attn_kernel(
    const unsigned short* __restrict__ Qh, const unsigned short* __restrict__ Kh,
    const unsigned short* __restrict__ Vh, const int* __restrict__ mask,
    unsigned short* __restrict__ ctx) {
  __shared__ unsigned short Ks[32 * 66];      // [key][d]  stride 66 (TDM pad)
  __shared__ unsigned short Vs[64 * 34];      // [d][key]  stride 34 (transposed)
  __shared__ unsigned short Ps[8 * 16 * 34];  // per-wave P scratch [row][key]

  const int tid  = threadIdx.x;
  const int wid  = tid >> 5;
  const int lane = tid & 31;
  const int hi   = lane >> 4;
  const int ln   = lane & 15;
  const int bh   = blockIdx.y;  // b*H + h
  const int b    = bh >> 4;
  const int h    = bh & 15;
  const int q0   = blockIdx.x * 128 + wid * 16;

  const size_t headBase = (size_t)bh * S_ * HD_;
#if USE_TDM
  const unsigned ksOff = (unsigned)(size_t)(void*)Ks;  // LDS byte address
#endif

  // Q fragments: 16x64 as two 16x32 A-fragments, kept in registers.
  FragA qf[2];
#pragma unroll
  for (int c = 0; c < 2; ++c) {
    const unsigned short* qrow = Qh + headBase + (size_t)(q0 + ln) * HD_ + c * 32;
#pragma unroll
    for (int j = 0; j < 8; ++j)
      qf[c].u[j] = *(const unsigned int*)(qrow + kkOf(j, hi));
  }

  const v8f vz = {0.f, 0.f, 0.f, 0.f, 0.f, 0.f, 0.f, 0.f};
  v8f o[4] = {vz, vz, vz, vz};  // 16x64 output accumulator (4 d-tiles)
  float mrun[8], lrun[8];
#pragma unroll
  for (int r = 0; r < 8; ++r) { mrun[r] = -3.0e38f; lrun[r] = 0.f; }

  const float scale = 1.0f / (8.0f + 1e-8f);  // 1/(sqrt(64)+1e-8)

  for (int kt = 0; kt < S_; kt += 32) {
#if USE_TDM
    if (wid == 0)  // one wave drives the tensor DMA for the K tile
      tdm_load_k_tile(ksOff, Kh + headBase + (size_t)kt * HD_);
#endif
    if (kt + 32 < S_)
      __builtin_prefetch(Vh + headBase + (size_t)(kt + 32 + (tid >> 3)) * HD_, 0, 1);
    // V tile loaded transposed by the threads (TDM cannot transpose).
#pragma unroll
    for (int i = 0; i < 8; ++i) {
      int idx = tid + i * 256;
      int key = idx >> 6, d = idx & 63;
      size_t g = headBase + (size_t)(kt + key) * HD_ + d;
#if !USE_TDM
      Ks[key * 66 + d] = Kh[g];
#endif
      Vs[d * 34 + key] = Vh[g];
    }
#if USE_TDM
    if (wid == 0) __builtin_amdgcn_s_wait_tensorcnt(0);
#endif
    __syncthreads();

    // Scores: two 16x16 C tiles (keys kt..+15, kt+16..+31), K-dim 64 = 2 chunks.
    v8f sc[2];
#pragma unroll
    for (int t = 0; t < 2; ++t) {
      v8f c = vz;
#pragma unroll
      for (int ch = 0; ch < 2; ++ch) {
        v16bf bf = load_frag_lds(Ks, 66, t * 16 + ln, hi, ch * 32);
        c = __builtin_amdgcn_wmma_f32_16x16x32_bf16(false, qf[ch].v, false, bf,
                                                    (short)0, c, false, false);
      }
      sc[t] = c;
    }

    // Scale + additive mask (-1e9 * mask[b, key]); per-lane key column.
    float s0[8], s1[8];
    {
      float p0 = (float)mask[b * S_ + kt + ln] * -1e9f;
      float p1 = (float)mask[b * S_ + kt + 16 + ln] * -1e9f;
#pragma unroll
      for (int r = 0; r < 8; ++r) {
        s0[r] = sc[0][r] * scale + p0;
        s1[r] = sc[1][r] * scale + p1;
      }
    }

    // Online softmax: row reductions across the 16 lanes of each half-wave.
    float fac[8];
#pragma unroll
    for (int r = 0; r < 8; ++r) {
      float m = fmaxf(s0[r], s1[r]);
#pragma unroll
      for (int x = 1; x < 16; x <<= 1) m = fmaxf(m, __shfl_xor(m, x, 32));
      float mnew = fmaxf(mrun[r], m);
      fac[r] = __expf(mrun[r] - mnew);
      float e0 = __expf(s0[r] - mnew);
      float e1 = __expf(s1[r] - mnew);
      s0[r] = e0; s1[r] = e1;
      float sum = e0 + e1;
#pragma unroll
      for (int x = 1; x < 16; x <<= 1) sum += __shfl_xor(sum, x, 32);
      lrun[r] = lrun[r] * fac[r] + sum;
      mrun[r] = mnew;
    }

    // Rescale running output.
#pragma unroll
    for (int dt = 0; dt < 4; ++dt)
#pragma unroll
      for (int r = 0; r < 8; ++r) o[dt][r] *= fac[r];

    // P: C-layout -> A-layout via per-wave LDS bounce (bf16).
    unsigned short* myP = Ps + wid * (16 * 34);
#pragma unroll
    for (int r = 0; r < 8; ++r) {
      myP[(r + hi * 8) * 34 + ln]      = f2bf(s0[r]);
      myP[(r + hi * 8) * 34 + 16 + ln] = f2bf(s1[r]);
    }
    asm volatile("" ::: "memory");  // keep program order; LDS is in-order per wave
    FragA pf;
#pragma unroll
    for (int j = 0; j < 8; ++j)
      pf.u[j] = *(const unsigned int*)(myP + ln * 34 + kkOf(j, hi));

    // O += P (16x32) * V (32x64), 4 d-tiles.
#pragma unroll
    for (int dt = 0; dt < 4; ++dt) {
      v16bf vb = load_frag_lds(Vs, 34, dt * 16 + ln, hi, 0);
      o[dt] = __builtin_amdgcn_wmma_f32_16x16x32_bf16(false, pf.v, false, vb,
                                                      (short)0, o[dt], false, false);
    }
    __syncthreads();
  }

  // Normalize and write merged-head ctx (bf16, [B*S, D] row-major).
#pragma unroll
  for (int dt = 0; dt < 4; ++dt)
#pragma unroll
    for (int r = 0; r < 8; ++r) {
      int q = q0 + r + hi * 8;
      int d = dt * 16 + ln;
      ctx[(size_t)(b * S_ + q) * D_ + h * HD_ + d] = f2bf(o[dt][r] / lrun[r]);
    }
}

// ---------------------------------------------------------------------------
extern "C" void kernel_launch(void* const* d_in, const int* in_sizes, int n_in,
                              void* d_out, int out_size, void* d_ws, size_t ws_size,
                              hipStream_t stream) {
  (void)in_sizes; (void)n_in; (void)out_size; (void)ws_size;
  const float* q    = (const float*)d_in[0];
  const float* k    = (const float*)d_in[1];
  const float* v    = (const float*)d_in[2];
  const int*   mask = (const int*)d_in[3];
  const float* Wq   = (const float*)d_in[4];
  const float* bq   = (const float*)d_in[5];
  const float* Wk   = (const float*)d_in[6];
  const float* bk   = (const float*)d_in[7];
  const float* Wv   = (const float*)d_in[8];
  const float* bv   = (const float*)d_in[9];
  const float* Wo   = (const float*)d_in[10];
  const float* bo   = (const float*)d_in[11];

  const size_t elems = (size_t)B_ * S_ * D_;  // 8M elements
  unsigned short* Qh  = (unsigned short*)d_ws;        // bf16 [B,H,S,HD]
  unsigned short* Kh  = Qh + elems;
  unsigned short* Vh  = Kh + elems;
  unsigned short* ctx = Vh + elems;                   // bf16 [B*S, D]

  dim3 ggrid(D_ / 64, (B_ * S_) / 128);  // 16 x 64 blocks
  gemm_kernel<float, true><<<ggrid, 256, 0, stream>>>(q, Wq, bq, Qh);
  gemm_kernel<float, true><<<ggrid, 256, 0, stream>>>(k, Wk, bk, Kh);
  gemm_kernel<float, true><<<ggrid, 256, 0, stream>>>(v, Wv, bv, Vh);

  dim3 agrid(S_ / 128, B_ * H_);         // 16 x 64 blocks
  flash_attn_kernel<<<agrid, 256, 0, stream>>>(Qh, Kh, Vh, mask, ctx);

  gemm_kernel<unsigned short, false><<<ggrid, 256, 0, stream>>>(ctx, Wo, bo, d_out);
}
